// SixLineAttention_46832323395753
// MI455X (gfx1250) — matrix operand order repaired and checked by reading, not compiled, not run on hardware
//
#include <hip/hip_runtime.h>
#include <hip/hip_bf16.h>

// SixLineAttention for MI455X (gfx1250), wave32, WMMA bf16 + async LDS DMA.
// The anchor bias lam[h]*dot(q, anchor_h) is constant across the key axis and
// softmax is shift-invariant per row => it cancels exactly; anchor inputs are
// legitimately unused.

typedef __attribute__((ext_vector_type(16))) __bf16 v16bf;
typedef __attribute__((ext_vector_type(8)))  __bf16 v8bf;
typedef __attribute__((ext_vector_type(8)))  float  v8f;

#define C_B   2
#define C_T   4096
#define C_D   768
#define C_H   6
#define C_HD  128
#define C_M   (C_B * C_T)   // 8192
#define C_NQKV (3 * C_D)    // 2304

static __device__ __forceinline__ v16bf cat8(v8bf lo, v8bf hi) {
  return __builtin_shufflevector(lo, hi, 0,1,2,3,4,5,6,7,8,9,10,11,12,13,14,15);
}

// ---- CDNA5 async global->LDS DMA (ASYNCcnt-tracked, no data VGPRs) --------
static __device__ __forceinline__ void async_ld_b128(__bf16* lds_dst,
                                                     const __bf16* gsrc) {
  uint32_t l = (uint32_t)(uintptr_t)(__attribute__((address_space(3))) __bf16*)lds_dst;
  asm volatile("global_load_async_to_lds_b128 %0, %1, off"
               :: "v"(l), "v"((uint64_t)(uintptr_t)gsrc)
               : "memory");
}
#define S_WAIT_ASYNCCNT(N) asm volatile("s_wait_asynccnt %0" :: "n"(N) : "memory")

// ---------------------------------------------------------------- prep ----
__global__ void cvt_x_kernel(const float* __restrict__ x, __bf16* __restrict__ xb, int n) {
  int i = blockIdx.x * blockDim.x + threadIdx.x;
  if (i < n) xb[i] = (__bf16)x[i];
}

__global__ void cvt_w_kernel(const float* __restrict__ Wq, const float* __restrict__ Wk,
                             const float* __restrict__ Wv, const float* __restrict__ Wo,
                             __bf16* __restrict__ wqkv, __bf16* __restrict__ wob) {
  int i = blockIdx.x * blockDim.x + threadIdx.x;
  if (i < C_NQKV * C_D) {
    int r = i / C_D, c = i % C_D;
    float v = (r < C_D) ? Wq[r * C_D + c]
            : (r < 2 * C_D) ? Wk[(r - C_D) * C_D + c]
            : Wv[(r - 2 * C_D) * C_D + c];
    wqkv[i] = (__bf16)v;
  }
  if (i < C_D * C_D) wob[i] = (__bf16)Wo[i];
}

// ------------------------------------------------------------ QKV GEMM ----
// y = x @ W^T  (M=8192, N=2304, K=768).  A(xb) row-major [m][k] bf16,
// B(wqkv) row-major [n][k] bf16.  128x128 tile, 8 waves (4x2), each wave
// 32x64 = 2x4 WMMA tiles.  Double-buffered async LDS staging.
__global__ __launch_bounds__(256)
void qkv_gemm_kernel(const __bf16* __restrict__ xb, const __bf16* __restrict__ wqkv,
                     __bf16* __restrict__ q_ws, __bf16* __restrict__ k_ws,
                     __bf16* __restrict__ vt_ws) {
  __shared__ __bf16 As[2][128 * 32];
  __shared__ __bf16 Bs[2][128 * 32];
  const int tid  = threadIdx.x;
  const int lane = tid & 31;
  const int wid  = tid >> 5;
  const int wm   = wid >> 1;      // 0..3
  const int wn   = wid & 1;       // 0..1
  const int m0   = blockIdx.y * 128;
  const int n0   = blockIdx.x * 128;
  const int am    = lane & 15;
  const int ahalf = lane >> 4;
  const int lr = tid >> 1;          // 0..127
  const int lc = (tid & 1) * 16;    // 0 or 16

  v8f acc[2][4];
#pragma unroll
  for (int mt = 0; mt < 2; mt++)
#pragma unroll
    for (int nt = 0; nt < 4; nt++) acc[mt][nt] = (v8f)0.0f;

  auto stage = [&](int buf, int k0) {  // 4 async b128 per wave
    const __bf16* ga = &xb  [(size_t)(m0 + lr) * C_D + k0 + lc];
    const __bf16* gb = &wqkv[(size_t)(n0 + lr) * C_D + k0 + lc];
    __bf16* la = &As[buf][lr * 32 + lc];
    __bf16* lb = &Bs[buf][lr * 32 + lc];
    async_ld_b128(la,     ga);
    async_ld_b128(la + 8, ga + 8);
    async_ld_b128(lb,     gb);
    async_ld_b128(lb + 8, gb + 8);
  };

  stage(0, 0);
  const int NIT = C_D / 32;  // 24
  for (int it = 0; it < NIT; it++) {
    const int cur = it & 1;
    if (it + 1 < NIT) {
      stage(cur ^ 1, (it + 1) * 32);
      S_WAIT_ASYNCCNT(4);    // in-order => current buffer's loads complete
    } else {
      S_WAIT_ASYNCCNT(0);
    }
    __syncthreads();

    v16bf afr[2];
#pragma unroll
    for (int mt = 0; mt < 2; mt++) {
      const __bf16* ap = &As[cur][(wm * 32 + mt * 16 + am) * 32];
      v8bf lo = *(const v8bf*)&ap[ahalf * 16];
      v8bf hi = *(const v8bf*)&ap[ahalf * 16 + 8];
      afr[mt] = cat8(lo, hi);
    }
#pragma unroll
    for (int nt = 0; nt < 4; nt++) {
      const __bf16* bp = &Bs[cur][(wn * 64 + nt * 16 + am) * 32 + ahalf * 16];
      v16bf bfr = *(const v16bf*)bp;
#pragma unroll
      for (int mt = 0; mt < 2; mt++)
        acc[mt][nt] = __builtin_amdgcn_wmma_f32_16x16x32_bf16(
            false, afr[mt], false, bfr, (short)0, acc[mt][nt], false, false);
    }
    __syncthreads();  // done reading 'cur' before it is refilled at it+2
  }

  // Epilogue: route to Q / K (row-major per head) or V^T ([hd][t] per head).
#pragma unroll
  for (int mt = 0; mt < 2; mt++)
#pragma unroll
    for (int nt = 0; nt < 4; nt++) {
      int n   = n0 + wn * 64 + nt * 16 + am;
      int sel = n / C_D;
      int nn  = n % C_D;
      int h   = nn / C_HD;
      int hd  = nn % C_HD;
#pragma unroll
      for (int v = 0; v < 8; v++) {
        int m = m0 + wm * 32 + mt * 16 + v + 8 * ahalf;
        int b = m / C_T, t = m % C_T;
        __bf16 val = (__bf16)acc[mt][nt][v];
        size_t bh = (size_t)(b * C_H + h);
        if (sel == 0)      q_ws [(bh * C_T + t) * C_HD + hd] = val;
        else if (sel == 1) k_ws [(bh * C_T + t) * C_HD + hd] = val;
        else               vt_ws[(bh * C_HD + hd) * C_T + t] = val;
      }
    }
}

// ------------------------------------------------------ flash attention ----
// One workgroup = (b,h) x 64 query rows; 4 waves x 16 rows. K-blocks of 32
// keys (one WMMA K-depth). Online softmax in the C-fragment layout.
// Double-buffered async LDS staging of K/V blocks.
__global__ __launch_bounds__(128)
void attn_kernel(const __bf16* __restrict__ q_ws, const __bf16* __restrict__ k_ws,
                 const __bf16* __restrict__ vt_ws, __bf16* __restrict__ ao_ws) {
  __shared__ __bf16 Ks[2][32 * 128];   // [tk][hd]
  __shared__ __bf16 Vs[2][128 * 32];   // [hd][tk]
  __shared__ __bf16 Ps[4 * 16 * 32];   // per-wave P staging [m][tk]

  const int bh  = blockIdx.y;            // 0..11
  const int q0  = blockIdx.x * 64;
  const int tid = threadIdx.x;
  const int lane = tid & 31;
  const int wid  = tid >> 5;             // 0..3
  const int am    = lane & 15;
  const int ahalf = lane >> 4;

  const __bf16* Qh = q_ws  + (size_t)bh * C_T * C_HD;
  const __bf16* Kh = k_ws  + (size_t)bh * C_T * C_HD;
  const __bf16* Vh = vt_ws + (size_t)bh * C_HD * C_T;

  // Q fragments for this wave's 16 rows, all 128 hd -> 4 K-steps.
  v16bf qf[4];
#pragma unroll
  for (int ks = 0; ks < 4; ks++) {
    const __bf16* qp = &Qh[(size_t)(q0 + wid * 16 + am) * C_HD + ks * 32];
    v8bf lo = *(const v8bf*)&qp[ahalf * 16];
    v8bf hi = *(const v8bf*)&qp[ahalf * 16 + 8];
    qf[ks] = cat8(lo, hi);
  }

  v8f O[8];
#pragma unroll
  for (int nt = 0; nt < 8; nt++) O[nt] = (v8f)0.0f;
  float m_r[8], l_r[8];
#pragma unroll
  for (int v = 0; v < 8; v++) { m_r[v] = -1e30f; l_r[v] = 0.0f; }
  const float scale = 0.08838834764831845f;  // 1/sqrt(128)

  const int kr = tid >> 2;           // 0..31 (K rows)
  const int kc = (tid & 3) * 32;     // 0,32,64,96 (hd cols)

  auto stage_kv = [&](int buf, int kb) {  // 8 async b128 per wave
    const __bf16* gk = &Kh[(size_t)(kb + kr) * C_HD + kc];
    const __bf16* gv = &Vh[(size_t)tid * C_T + kb];
    __bf16* lk = &Ks[buf][kr * 128 + kc];
    __bf16* lv = &Vs[buf][tid * 32];
    async_ld_b128(lk,      gk);
    async_ld_b128(lk + 8,  gk + 8);
    async_ld_b128(lk + 16, gk + 16);
    async_ld_b128(lk + 24, gk + 24);
    async_ld_b128(lv,      gv);
    async_ld_b128(lv + 8,  gv + 8);
    async_ld_b128(lv + 16, gv + 16);
    async_ld_b128(lv + 24, gv + 24);
  };

  stage_kv(0, 0);
  const int NBLK = C_T / 32;  // 128
  for (int it = 0; it < NBLK; it++) {
    const int cur = it & 1;
    if (it + 1 < NBLK) {
      stage_kv(cur ^ 1, (it + 1) * 32);
      S_WAIT_ASYNCCNT(8);   // in-order => current block's loads complete
    } else {
      S_WAIT_ASYNCCNT(0);
    }
    __syncthreads();

    // S = Q K^T for this wave's 16 rows x 32 keys.
    v8f S[2];
    S[0] = (v8f)0.0f; S[1] = (v8f)0.0f;
#pragma unroll
    for (int nt = 0; nt < 2; nt++)
#pragma unroll
      for (int ks = 0; ks < 4; ks++) {
        const __bf16* bp = &Ks[cur][(nt * 16 + am) * 128 + ks * 32 + ahalf * 16];
        v16bf bfr = *(const v16bf*)bp;
        S[nt] = __builtin_amdgcn_wmma_f32_16x16x32_bf16(
            false, qf[ks], false, bfr, (short)0, S[nt], false, false);
      }

    // Online softmax (row = vgpr slot, cols = 16 lanes of this half).
    float alpha[8];
#pragma unroll
    for (int v = 0; v < 8; v++) {
      float s0 = S[0][v] * scale, s1 = S[1][v] * scale;
      float mx = fmaxf(s0, s1);
      mx = fmaxf(mx, __shfl_xor(mx, 1, 32));
      mx = fmaxf(mx, __shfl_xor(mx, 2, 32));
      mx = fmaxf(mx, __shfl_xor(mx, 4, 32));
      mx = fmaxf(mx, __shfl_xor(mx, 8, 32));
      float mn = fmaxf(m_r[v], mx);
      alpha[v] = __expf(m_r[v] - mn);
      m_r[v] = mn;
      float p0 = __expf(s0 - mn);
      float p1 = __expf(s1 - mn);
      S[0][v] = p0; S[1][v] = p1;
      float rs = p0 + p1;
      rs += __shfl_xor(rs, 1, 32);
      rs += __shfl_xor(rs, 2, 32);
      rs += __shfl_xor(rs, 4, 32);
      rs += __shfl_xor(rs, 8, 32);
      l_r[v] = l_r[v] * alpha[v] + rs;
    }

    // Stage P (C-frag layout -> A-frag layout) through per-wave LDS.
    __bf16* Pw = &Ps[wid * 16 * 32];
#pragma unroll
    for (int v = 0; v < 8; v++) {
      int mrow = v + 8 * ahalf;
      Pw[mrow * 32 + am]      = (__bf16)S[0][v];
      Pw[mrow * 32 + 16 + am] = (__bf16)S[1][v];
    }
#pragma unroll
    for (int nt = 0; nt < 8; nt++)
#pragma unroll
      for (int v = 0; v < 8; v++) O[nt][v] *= alpha[v];

    v16bf pf;
    {
      const __bf16* pp = &Pw[am * 32];
      v8bf lo = *(const v8bf*)&pp[ahalf * 16];
      v8bf hi = *(const v8bf*)&pp[ahalf * 16 + 8];
      pf = cat8(lo, hi);
    }
#pragma unroll
    for (int nt = 0; nt < 8; nt++) {
      const __bf16* vp = &Vs[cur][(nt * 16 + am) * 32 + ahalf * 16];
      v16bf vfr = *(const v16bf*)vp;
      O[nt] = __builtin_amdgcn_wmma_f32_16x16x32_bf16(
          false, pf, false, vfr, (short)0, O[nt], false, false);
    }
    __syncthreads();  // done reading 'cur' before it is refilled at it+2
  }

  // Normalize and store bf16 attention output row-major [b*T + t][h*HD + hd].
  const int b = bh / C_H, h = bh % C_H;
#pragma unroll
  for (int v = 0; v < 8; v++) {
    float inv = 1.0f / l_r[v];
    int t = q0 + wid * 16 + v + 8 * ahalf;
#pragma unroll
    for (int nt = 0; nt < 8; nt++) {
      int hd = nt * 16 + am;
      ao_ws[((size_t)(b * C_T + t)) * C_D + h * C_HD + hd] = (__bf16)(O[nt][v] * inv);
    }
  }
}

// --------------------------------------------------------- output GEMM ----
// out = ao @ Wo^T  (M=8192, N=768, K=768), f32 result.
__global__ __launch_bounds__(256)
void out_gemm_kernel(const __bf16* __restrict__ ao, const __bf16* __restrict__ wob,
                     float* __restrict__ out) {
  __shared__ __bf16 As[2][128 * 32];
  __shared__ __bf16 Bs[2][128 * 32];
  const int tid  = threadIdx.x;
  const int lane = tid & 31;
  const int wid  = tid >> 5;
  const int wm   = wid >> 1;
  const int wn   = wid & 1;
  const int m0   = blockIdx.y * 128;
  const int n0   = blockIdx.x * 128;
  const int am    = lane & 15;
  const int ahalf = lane >> 4;
  const int lr = tid >> 1;
  const int lc = (tid & 1) * 16;

  v8f acc[2][4];
#pragma unroll
  for (int mt = 0; mt < 2; mt++)
#pragma unroll
    for (int nt = 0; nt < 4; nt++) acc[mt][nt] = (v8f)0.0f;

  auto stage = [&](int buf, int k0) {
    const __bf16* ga = &ao [(size_t)(m0 + lr) * C_D + k0 + lc];
    const __bf16* gb = &wob[(size_t)(n0 + lr) * C_D + k0 + lc];
    __bf16* la = &As[buf][lr * 32 + lc];
    __bf16* lb = &Bs[buf][lr * 32 + lc];
    async_ld_b128(la,     ga);
    async_ld_b128(la + 8, ga + 8);
    async_ld_b128(lb,     gb);
    async_ld_b128(lb + 8, gb + 8);
  };

  stage(0, 0);
  const int NIT = C_D / 32;
  for (int it = 0; it < NIT; it++) {
    const int cur = it & 1;
    if (it + 1 < NIT) {
      stage(cur ^ 1, (it + 1) * 32);
      S_WAIT_ASYNCCNT(4);
    } else {
      S_WAIT_ASYNCCNT(0);
    }
    __syncthreads();

    v16bf afr[2];
#pragma unroll
    for (int mt = 0; mt < 2; mt++) {
      const __bf16* ap = &As[cur][(wm * 32 + mt * 16 + am) * 32];
      v8bf lo = *(const v8bf*)&ap[ahalf * 16];
      v8bf hi = *(const v8bf*)&ap[ahalf * 16 + 8];
      afr[mt] = cat8(lo, hi);
    }
#pragma unroll
    for (int nt = 0; nt < 4; nt++) {
      const __bf16* bp = &Bs[cur][(wn * 64 + nt * 16 + am) * 32 + ahalf * 16];
      v16bf bfr = *(const v16bf*)bp;
#pragma unroll
      for (int mt = 0; mt < 2; mt++)
        acc[mt][nt] = __builtin_amdgcn_wmma_f32_16x16x32_bf16(
            false, afr[mt], false, bfr, (short)0, acc[mt][nt], false, false);
    }
    __syncthreads();
  }

#pragma unroll
  for (int mt = 0; mt < 2; mt++)
#pragma unroll
    for (int nt = 0; nt < 4; nt++) {
      int n = n0 + wn * 64 + nt * 16 + am;
#pragma unroll
      for (int v = 0; v < 8; v++) {
        int m = m0 + wm * 32 + mt * 16 + v + 8 * ahalf;
        out[(size_t)m * C_D + n] = acc[mt][nt][v];
      }
    }
}

// ------------------------------------------------------------- launch ----
extern "C" void kernel_launch(void* const* d_in, const int* in_sizes, int n_in,
                              void* d_out, int out_size, void* d_ws, size_t ws_size,
                              hipStream_t stream) {
  const float* x  = (const float*)d_in[0];
  const float* Wq = (const float*)d_in[1];
  const float* Wk = (const float*)d_in[2];
  const float* Wv = (const float*)d_in[3];
  const float* Wo = (const float*)d_in[4];
  // d_in[5..7] (anchor_w, anchors, lam): softmax shift-invariance => unused.
  (void)in_sizes; (void)n_in; (void)out_size; (void)ws_size;

  size_t off = 0;
  auto carve = [&](size_t bytes) -> void* {
    void* p = (char*)d_ws + off;
    off += (bytes + 255) & ~(size_t)255;
    return p;
  };
  __bf16* xb    = (__bf16*)carve((size_t)C_M * C_D * 2);
  __bf16* wqkv  = (__bf16*)carve((size_t)C_NQKV * C_D * 2);
  __bf16* wob   = (__bf16*)carve((size_t)C_D * C_D * 2);
  __bf16* q_ws  = (__bf16*)carve((size_t)C_B * C_H * C_T * C_HD * 2);
  __bf16* k_ws  = (__bf16*)carve((size_t)C_B * C_H * C_T * C_HD * 2);
  __bf16* vt_ws = (__bf16*)carve((size_t)C_B * C_H * C_T * C_HD * 2);
  __bf16* ao_ws = (__bf16*)carve((size_t)C_M * C_D * 2);

  {
    int n = C_M * C_D;
    cvt_x_kernel<<<(n + 255) / 256, 256, 0, stream>>>(x, xb, n);
  }
  {
    int n = C_NQKV * C_D;
    cvt_w_kernel<<<(n + 255) / 256, 256, 0, stream>>>(Wq, Wk, Wv, Wo, wqkv, wob);
  }
  {
    dim3 grid(C_NQKV / 128, C_M / 128);  // (18, 64)
    qkv_gemm_kernel<<<grid, 256, 0, stream>>>(xb, wqkv, q_ws, k_ws, vt_ws);
  }
  {
    dim3 grid(C_T / 64, C_B * C_H);      // (64, 12)
    attn_kernel<<<grid, 128, 0, stream>>>(q_ws, k_ws, vt_ws, ao_ws);
  }
  {
    dim3 grid(C_D / 128, C_M / 128);     // (6, 64)
    out_gemm_kernel<<<grid, 256, 0, stream>>>(ao_ws, wob, (float*)d_out);
  }
}
